// MiniYoloDetector_89661737271765
// MI455X (gfx1250) — compile-verified, matrix operand order checked by reading, not run on hardware
//
#include <hip/hip_runtime.h>
#include <stdint.h>

// ---------------- problem constants ----------------
#define BB      32
#define CC      80
#define HSZ     80
#define WSZ     80
#define NCELL   (HSZ*WSZ)        // 6400
#define NSCORE  (NCELL*CC)       // 512000
#define IMGF    640.0f
#define CONF    0.005f
#define NMST    0.6f
#define TOPK    300
#define MAXDET  100

// ---------------- selection constants ----------------
#define BINS     4096
#define BINSHIFT 18
#define CAP      4096
#define HIREG    512             // region [0,512) = bins strictly above T (<300 guaranteed)
#define EQREG    (CAP - HIREG)   // region [512,4096) = tie-bin entries
#define SEGSZ    8192
#define SEGS     63              // ceil(512000/8192)
#define CHUNK    2048            // async LDS staging chunk (8 KB)

typedef _Float16 v16h __attribute__((ext_vector_type(16)));
typedef float    v8f  __attribute__((ext_vector_type(8)));

// CDNA5 async global->LDS path (ASYNCcnt), guarded so either toolchain compiles.
#if defined(__HIP_DEVICE_COMPILE__) && \
    __has_builtin(__builtin_amdgcn_global_load_async_to_lds_b32) && \
    __has_builtin(__builtin_amdgcn_s_wait_asynccnt)
#define HAVE_ASYNC_LDS 1
typedef __attribute__((address_space(1))) int gas_i32;   // global (AS1) word
typedef __attribute__((address_space(3))) int las_i32;   // LDS (AS3) word
#else
#define HAVE_ASYNC_LDS 0
#endif

__device__ __forceinline__ float sigm(float x) { return 1.0f / (1.0f + expf(-x)); }

// ---------------- K0: zero counters / hist / output ----------------
__global__ void init_kernel(unsigned* __restrict__ hist,
                            unsigned* __restrict__ cntHi,
                            unsigned* __restrict__ cntEq,
                            unsigned* __restrict__ binT,
                            float* __restrict__ dout, int outN) {
    int i = blockIdx.x * blockDim.x + threadIdx.x;
    if (i < BB * BINS) hist[i] = 0u;
    if (i < BB) { cntHi[i] = 0u; cntEq[i] = 0u; binT[i] = 0u; }
    if (i < outN) dout[i] = 0.0f;
}

// ---------------- K1: decode boxes + masked scores (WMMA softmax-sum) -------
// 256 threads = 8 waves; each wave owns 16 cells; block owns 128 cells.
__global__ __launch_bounds__(256)
void decode_score_kernel(const float* __restrict__ p,
                         float* __restrict__ boxes,
                         float* __restrict__ smask) {
    __shared__ float s_exp[8][16][96];  // f32 exp values, K zero-padded 80..95
    __shared__ float s_sum[8][16];
    __shared__ float s_obj[128];

    const int tid = threadIdx.x;
    const int b = blockIdx.x / (NCELL / 128);
    const int cellBase = (blockIdx.x % (NCELL / 128)) * 128;
    const float* pimg = p + (size_t)b * (5 + CC) * NCELL;

    // ---- phase 1: boxes + objectness (128 cells, one thread each) ----
    if (tid < 128) {
        int cell = cellBase + tid;
        float tx = pimg[0 * NCELL + cell];
        float ty = pimg[1 * NCELL + cell];
        float tw = pimg[2 * NCELL + cell];
        float th = pimg[3 * NCELL + cell];
        float to = pimg[4 * NCELL + cell];
        int gy = cell / WSZ, gx = cell % WSZ;
        float cx = (sigm(tx) + (float)gx) * 8.0f;
        float cy = (sigm(ty) + (float)gy) * 8.0f;
        float w  = expf(tw) * 8.0f;
        float h  = expf(th) * 8.0f;
        float x1 = fminf(fmaxf(cx - 0.5f * w, 0.0f), IMGF - 1.0f);
        float y1 = fminf(fmaxf(cy - 0.5f * h, 0.0f), IMGF - 1.0f);
        float x2 = fminf(fmaxf(cx + 0.5f * w, 0.0f), IMGF - 1.0f);
        float y2 = fminf(fmaxf(cy + 0.5f * h, 0.0f), IMGF - 1.0f);
        float4 bx; bx.x = x1; bx.y = y1; bx.z = x2; bx.w = y2;
        ((float4*)boxes)[(size_t)b * NCELL + cell] = bx;
        s_obj[tid] = sigm(to);
    }

    // ---- phase 2: exp of class logits into LDS ----
    const int w_id = tid >> 5;
    const int lane = tid & 31;
    const int row  = lane & 15;   // cell row within the wave's 16x tile (M)
    const int half = lane >> 4;   // 0: classes 0..39, 1: classes 40..79
    const int cell = cellBase + w_id * 16 + row;
    for (int j = 0; j < 40; ++j) {
        int c = half * 40 + j;
        s_exp[w_id][row][c] = expf(pimg[(size_t)(5 + c) * NCELL + cell]);
    }
    if (half == 1) {
        for (int j = 0; j < 16; ++j) s_exp[w_id][row][80 + j] = 0.0f;
    }
    __syncthreads();

    // ---- phase 3: softmax denominators via V_WMMA_F32_16X16X32_F16 ----
    // D = A(16x32 f16 exps) x B(32x16 ones) accumulated over K = 96.
    // 16-bit A layout: lanes 0-15 hold K {0..7,16..23}; lanes 16-31 hold
    // K {8..15,24..31} of row M = lane%16.
    v16h ones;
    #pragma unroll
    for (int i = 0; i < 16; ++i) ones[i] = (_Float16)1.0f;
    v8f acc = {};
    #pragma unroll
    for (int kb = 0; kb < 3; ++kb) {
        v16h afrag;
        #pragma unroll
        for (int i = 0; i < 16; ++i) {
            int kk = (half == 0) ? ((i < 8) ? i : i + 8)
                                 : ((i < 8) ? i + 8 : i + 16);
            afrag[i] = (_Float16)s_exp[w_id][row][kb * 32 + kk];
        }
        acc = __builtin_amdgcn_wmma_f32_16x16x32_f16(
                  false, afrag, false, ones, (short)0, acc, false, false);
    }
    // C/D f32 layout: lanes 0-15 hold rows 0..7 (VGPR r = row r),
    // lanes 16-31 hold rows 8..15. One lane per half publishes the sums.
    if (lane == 0 || lane == 16) {
        #pragma unroll
        for (int r = 0; r < 8; ++r) s_sum[w_id][half * 8 + r] = acc[r];
    }
    __syncthreads();

    // ---- phase 4: masked scores (thresholded) ----
    float inv = 1.0f / s_sum[w_id][row];
    float obj = s_obj[w_id * 16 + row];
    size_t base = ((size_t)b * NCELL + (size_t)cell) * CC;
    for (int j = 0; j < 40; ++j) {
        int c = half * 40 + j;
        float s = obj * s_exp[w_id][row][c] * inv;
        smask[base + c] = (s > CONF) ? s : 0.0f;
    }
}

// ---------------- K2: per-image 4096-bin histogram of score bits ------------
// Streams smask through LDS via GLOBAL_LOAD_ASYNC_TO_LDS_B32 (ASYNCcnt path);
// each thread consumes only the LDS slots it issued, so a wave-local
// s_wait_asynccnt(0) is sufficient before the reads.
__global__ __launch_bounds__(256)
void hist_kernel(const float* __restrict__ smask, unsigned* __restrict__ hist) {
    __shared__ unsigned lh[BINS];
    __shared__ float tile[CHUNK];
    const int tid = threadIdx.x;
    const int b = blockIdx.x / SEGS;
    const int seg = blockIdx.x % SEGS;
    for (int i = tid; i < BINS; i += 256) lh[i] = 0u;
    __syncthreads();
    const int base = seg * SEGSZ;
    const float* src = smask + (size_t)b * NSCORE;

    for (int chunk = 0; chunk < SEGSZ / CHUNK; ++chunk) {
        const int cbase = base + chunk * CHUNK;
        // prefetch the next chunk into L2 while this one is consumed
        if (cbase + CHUNK < NSCORE)
            __builtin_prefetch(src + cbase + CHUNK + tid, 0, 1);
        for (int s = 0; s < CHUNK; s += 256) {
            int idx = cbase + s + tid;
            if (idx < NSCORE) {
#if HAVE_ASYNC_LDS
                __builtin_amdgcn_global_load_async_to_lds_b32(
                    (gas_i32*)(src + idx), (las_i32*)&tile[s + tid], 0, 0);
#else
                tile[s + tid] = src[idx];
#endif
            }
        }
#if HAVE_ASYNC_LDS
        __builtin_amdgcn_s_wait_asynccnt(0);
#endif
        __syncthreads();
        for (int s = 0; s < CHUNK; s += 256) {
            int idx = cbase + s + tid;
            if (idx < NSCORE) {
                float v = tile[s + tid];
                if (v > 0.0f) {
                    int bin = (int)(__float_as_uint(v) >> BINSHIFT);  // monotonic
                    atomicAdd(&lh[bin], 1u);
                }
            }
        }
        __syncthreads();
    }
    for (int i = tid; i < BINS; i += 256)
        if (lh[i]) atomicAdd(&hist[b * BINS + i], lh[i]);
}

// ---------------- K3: radix-select the top-300 threshold bin ----------------
__global__ void thresh_kernel(const unsigned* __restrict__ hist,
                              unsigned* __restrict__ binT) {
    int b = blockIdx.x;
    if (threadIdx.x == 0) {
        unsigned cum = 0; int T = 0;
        for (int bin = BINS - 1; bin >= 0; --bin) {
            cum += hist[b * BINS + bin];
            if (cum >= TOPK) { T = bin; break; }
        }
        binT[b] = (unsigned)T;
    }
}

// ---------------- K4: fused single-pass gather --------------------------
// bins > T  -> region [0,HIREG)   (population guaranteed < TOPK, never dropped)
// bin == T  -> region [HIREG,CAP) (tie bin, capped)
__global__ __launch_bounds__(256)
void gather_kernel(const float* __restrict__ smask,
                   const unsigned* __restrict__ binT,
                   unsigned* __restrict__ cntHi, unsigned* __restrict__ cntEq,
                   float* __restrict__ candS, int* __restrict__ candI) {
    const int b = blockIdx.x / SEGS;
    const int seg = blockIdx.x % SEGS;
    const int T = (int)binT[b];
    const int base = seg * SEGSZ;
    const float* src = smask + (size_t)b * NSCORE;
    for (int i = threadIdx.x; i < SEGSZ; i += 256) {
        int idx = base + i;
        if (idx >= NSCORE) continue;
        if (i + 256 < SEGSZ) __builtin_prefetch(src + idx + 256, 0, 1);
        float s = src[idx];
        if (s > 0.0f) {
            int bin = (int)(__float_as_uint(s) >> BINSHIFT);
            if (bin > T) {
                unsigned pos = atomicAdd(&cntHi[b], 1u);
                if (pos < HIREG) {
                    candS[(size_t)b * CAP + pos] = s;
                    candI[(size_t)b * CAP + pos] = idx;   // == cell*C + cls
                }
            } else if (bin == T) {
                unsigned pos = atomicAdd(&cntEq[b], 1u);
                if (pos < EQREG) {
                    candS[(size_t)b * CAP + HIREG + pos] = s;
                    candI[(size_t)b * CAP + HIREG + pos] = idx;
                }
            }
        }
    }
}

// ---------------- K5: bitonic sort + NMS + final compaction -----------------
__global__ __launch_bounds__(512)
void sort_nms_kernel(const float* __restrict__ boxes,
                     const float* __restrict__ candSg,
                     const int* __restrict__ candIg,
                     const unsigned* __restrict__ cntHi,
                     const unsigned* __restrict__ cntEq,
                     float* __restrict__ dout) {
    __shared__ float cs[CAP];
    __shared__ int   ci[CAP];
    __shared__ float tS[TOPK];
    __shared__ float tB[TOPK][4];
    __shared__ float tOff[TOPK];
    __shared__ float tArea[TOPK];
    __shared__ int   tCls[TOPK];
    __shared__ int   tKeep[TOPK];

    const int b = blockIdx.x, tid = threadIdx.x;
    int hi = (int)cntHi[b]; if (hi > HIREG) hi = HIREG;
    int eq = (int)cntEq[b]; if (eq > EQREG) eq = EQREG;
    for (int i = tid; i < CAP; i += 512) {
        float s; int idx;
        if (i < hi)           { s = candSg[(size_t)b * CAP + i];
                                idx = candIg[(size_t)b * CAP + i]; }
        else if (i < hi + eq) { s = candSg[(size_t)b * CAP + HIREG + (i - hi)];
                                idx = candIg[(size_t)b * CAP + HIREG + (i - hi)]; }
        else                  { s = -1.0f; idx = 0x7fffffff; }
        cs[i] = s; ci[i] = idx;
    }
    __syncthreads();

    // bitonic sort, "best first" = (score desc, idx asc) — matches top_k ties
    for (int k = 2; k <= CAP; k <<= 1) {
        for (int j = k >> 1; j > 0; j >>= 1) {
            for (int i = tid; i < CAP; i += 512) {
                int ixj = i ^ j;
                if (ixj > i) {
                    float si = cs[i], sj = cs[ixj];
                    int   ii = ci[i], ij = ci[ixj];
                    bool jb = (sj > si) || (sj == si && ij < ii);  // ixj better
                    bool ib = (si > sj) || (si == sj && ii < ij);  // i better
                    bool asc = ((i & k) == 0);
                    bool dosw = asc ? jb : ib;
                    if (dosw) { cs[i] = sj; cs[ixj] = si; ci[i] = ij; ci[ixj] = ii; }
                }
            }
            __syncthreads();
        }
    }

    // top-300 setup
    if (tid < TOPK) {
        float s = cs[tid]; int idx = ci[tid];
        if (s < 0.0f) { s = 0.0f; idx = 0; }          // pad safety
        int cell = idx / CC, cls = idx % CC;
        float4 bx = ((const float4*)boxes)[(size_t)b * NCELL + cell];
        tS[tid] = s;
        tB[tid][0] = bx.x; tB[tid][1] = bx.y; tB[tid][2] = bx.z; tB[tid][3] = bx.w;
        tCls[tid] = cls;
        tOff[tid] = (float)cls * (2.0f * IMGF);
        tArea[tid] = fmaxf(bx.z - bx.x, 0.0f) * fmaxf(bx.w - bx.y, 0.0f);
        tKeep[tid] = (s > CONF) ? 1 : 0;
    }
    __syncthreads();

    // serial-i, parallel-j greedy suppression (class-offset boxes)
    for (int i = 0; i < TOPK; ++i) {
        __syncthreads();
        if (tid > i && tid < TOPK && tKeep[i] && tKeep[tid]) {
            float oi = tOff[i], oj = tOff[tid];
            float xx1 = fmaxf(tB[i][0] + oi, tB[tid][0] + oj);
            float yy1 = fmaxf(tB[i][1] + oi, tB[tid][1] + oj);
            float xx2 = fminf(tB[i][2] + oi, tB[tid][2] + oj);
            float yy2 = fminf(tB[i][3] + oi, tB[tid][3] + oj);
            float inter = fmaxf(xx2 - xx1, 0.0f) * fmaxf(yy2 - yy1, 0.0f);
            float iou = inter / (tArea[i] + tArea[tid] - inter + 1e-9f);
            if (iou > NMST) tKeep[tid] = 0;
        }
    }
    __syncthreads();

    // stable compaction == top_k(where(keep, s, 0), 100)
    if (tid == 0) {
        float* oB = dout + (size_t)b * MAXDET * 4;
        float* oS = dout + (size_t)BB * MAXDET * 4 + (size_t)b * MAXDET;
        float* oL = dout + (size_t)BB * MAXDET * 5 + (size_t)b * MAXDET;
        float* oK = dout + (size_t)BB * MAXDET * 6 + (size_t)b * MAXDET;
        int n = 0;
        for (int j = 0; j < TOPK && n < MAXDET; ++j) {
            if (tKeep[j] && tS[j] > 0.0f) {
                oB[n * 4 + 0] = tB[j][0]; oB[n * 4 + 1] = tB[j][1];
                oB[n * 4 + 2] = tB[j][2]; oB[n * 4 + 3] = tB[j][3];
                oS[n] = tS[j]; oL[n] = (float)tCls[j];
                oK[n] = (tS[j] > CONF) ? 1.0f : 0.0f;
                ++n;
            }
        }
        for (int j = 0; j < TOPK && n < MAXDET; ++j) {
            if (!(tKeep[j] && tS[j] > 0.0f)) {
                oB[n * 4 + 0] = tB[j][0]; oB[n * 4 + 1] = tB[j][1];
                oB[n * 4 + 2] = tB[j][2]; oB[n * 4 + 3] = tB[j][3];
                oS[n] = 0.0f; oL[n] = (float)tCls[j]; oK[n] = 0.0f;
                ++n;
            }
        }
    }
}

// ---------------- host-side launch ----------------
extern "C" void kernel_launch(void* const* d_in, const int* in_sizes, int n_in,
                              void* d_out, int out_size, void* d_ws, size_t ws_size,
                              hipStream_t stream) {
    (void)in_sizes; (void)n_in; (void)ws_size;
    const float* p = (const float*)d_in[0];
    float* dout = (float*)d_out;

    // workspace carve-up (256B aligned offsets)
    char* ws = (char*)d_ws;
    size_t off = 0;
    float*    boxes  = (float*)(ws + off);    off += (size_t)BB * NCELL * 4 * 4;  // 3.28 MB
    float*    smask  = (float*)(ws + off);    off += (size_t)BB * NSCORE * 4;     // 65.5 MB
    unsigned* hist   = (unsigned*)(ws + off); off += (size_t)BB * BINS * 4;       // 0.52 MB
    unsigned* binT   = (unsigned*)(ws + off); off += 256;
    unsigned* cntHi  = (unsigned*)(ws + off); off += 256;
    unsigned* cntEq  = (unsigned*)(ws + off); off += 256;
    float*    candS  = (float*)(ws + off);    off += (size_t)BB * CAP * 4;        // 0.52 MB
    int*      candI  = (int*)(ws + off);      off += (size_t)BB * CAP * 4;        // 0.52 MB

    // K0: zero hist/counters/output (ws+out are not re-poisoned between replays)
    init_kernel<<<(BB * BINS + 255) / 256, 256, 0, stream>>>(hist, cntHi, cntEq, binT, dout, out_size);

    // K1: decode + scores (WMMA softmax denominators)
    decode_score_kernel<<<BB * (NCELL / 128), 256, 0, stream>>>(p, boxes, smask);

    // K2: per-image histogram (async global->LDS staging)
    hist_kernel<<<BB * SEGS, 256, 0, stream>>>(smask, hist);

    // K3: threshold bin
    thresh_kernel<<<BB, 32, 0, stream>>>(hist, binT);

    // K4: fused single-pass gather (dual-region candidate buffer)
    gather_kernel<<<BB * SEGS, 256, 0, stream>>>(smask, binT, cntHi, cntEq, candS, candI);

    // K5: sort + NMS + finalize
    sort_nms_kernel<<<BB, 512, 0, stream>>>(boxes, candS, candI, cntHi, cntEq, dout);
}
